// GraphTransformer_53936199303599
// MI455X (gfx1250) — matrix operand order repaired and checked by reading, not compiled, not run on hardware
//
#include <hip/hip_runtime.h>

typedef __attribute__((ext_vector_type(16))) __bf16 bf16x16;
typedef __attribute__((ext_vector_type(8)))  __bf16 bf16x8;
typedef __attribute__((ext_vector_type(4)))  __bf16 bf16x4;
typedef __attribute__((ext_vector_type(8)))  float  f32x8;

#define LRELU 0.01f

__device__ __forceinline__ void atomicMaxF(float* addr, float val) {
  int* ia = (int*)addr;
  int old = __float_as_int(*addr);
  while (__int_as_float(old) < val) {
    int assumed = old;
    old = atomicCAS(ia, assumed, __float_as_int(val));
    if (old == assumed) break;
  }
}

// ---------------------------------------------------------------------------
// Tiled WMMA GEMM: C[M,N](ldc) = act(A[M,K] @ W[K,N] + bias)   (bf16 inputs,
// f32 accumulate).  mode bit0 = leaky-relu, bit1 = accumulate into existing C.
// Block = 128 threads (4 waves), tile 64x64, K-step 32.
// Fast staging: b128 global loads + packed cvt + b64 LDS stores.
// Boundary staging: branchless (clamped address + select-0), no exec masking.
// ---------------------------------------------------------------------------
__global__ __launch_bounds__(128)
void k_gemm(const float* __restrict__ A, const float* __restrict__ W,
            const float* __restrict__ bias, float* __restrict__ C,
            int M, int K, int N, int ldc, int mode) {
  __shared__ __attribute__((aligned(16))) __bf16 As[64 * 40];
  __shared__ __attribute__((aligned(16))) __bf16 Bs[64 * 40];
  const int t = threadIdx.x;
  const int wave = t >> 5, lane = t & 31;
  const int brow = blockIdx.x * 64;
  const int bcol = blockIdx.y * 64;
  const bool fullM = (brow + 64 <= M);
  f32x8 acc[4] = {};

  for (int k0 = 0; k0 < K; k0 += 32) {
    const bool fullK = (k0 + 32 <= K);
    // ---- stage A tile (64x32) ----
    if (fullK && fullM) {
#pragma unroll
      for (int i = 0; i < 4; ++i) {
        int e = (i * 128 + t) * 4;
        int r = e >> 5, kk = e & 31;
        float4 v4 = *(const float4*)&A[(long)(brow + r) * K + k0 + kk];
        bf16x4 h;
        h[0] = (__bf16)v4.x; h[1] = (__bf16)v4.y;
        h[2] = (__bf16)v4.z; h[3] = (__bf16)v4.w;
        *(bf16x4*)&As[r * 40 + kk] = h;
      }
    } else {
      // branchless boundary path: clamp address, select 0 for OOB
#pragma unroll
      for (int i = 0; i < 16; ++i) {
        int e = i * 128 + t;
        int r = e >> 5, kk = e & 31;
        int gr = brow + r, gk = k0 + kk;
        int grc = gr < M ? gr : (M - 1);
        int gkc = gk < K ? gk : (K - 1);
        float v = A[(long)grc * K + gkc];
        v = (gr < M && gk < K) ? v : 0.0f;
        As[r * 40 + kk] = (__bf16)v;
      }
    }
    // ---- stage B tile (32x64), transposed into LDS ----
    if (fullK) {
#pragma unroll
      for (int i = 0; i < 4; ++i) {
        int e = (i * 128 + t) * 4;
        int kk = e >> 6, n = e & 63;
        float4 v4 = *(const float4*)&W[(long)(k0 + kk) * N + bcol + n];
        Bs[(n + 0) * 40 + kk] = (__bf16)v4.x;
        Bs[(n + 1) * 40 + kk] = (__bf16)v4.y;
        Bs[(n + 2) * 40 + kk] = (__bf16)v4.z;
        Bs[(n + 3) * 40 + kk] = (__bf16)v4.w;
      }
    } else {
#pragma unroll
      for (int i = 0; i < 16; ++i) {
        int e = i * 128 + t;
        int kk = e >> 6, n = e & 63;
        int gk = k0 + kk;
        int gkc = gk < K ? gk : (K - 1);
        float v = W[(long)gkc * N + bcol + n];
        v = (gk < K) ? v : 0.0f;
        Bs[n * 40 + kk] = (__bf16)v;
      }
    }
    // speculative prefetch of the next A tile (global_prefetch_b8)
    if (k0 + 32 < K)
      __builtin_prefetch(&A[(long)(brow + (t >> 1)) * K + k0 + 32], 0, 1);
    __syncthreads();

    // A fragment: lane holds row m, K runs [kb..kb+7] and [kb+16..kb+23]
    const int m = wave * 16 + (lane & 15);
    const int akb = (lane < 16) ? 0 : 8;
    bf16x8 a0 = *(const bf16x8*)&As[m * 40 + akb];
    bf16x8 a1 = *(const bf16x8*)&As[m * 40 + akb + 16];
    bf16x16 af;
#pragma unroll
    for (int i = 0; i < 8; ++i) { af[i] = a0[i]; af[i + 8] = a1[i]; }

    const int bkb = (lane < 16) ? 0 : 8;
#pragma unroll
    for (int nb = 0; nb < 4; ++nb) {
      const int n = nb * 16 + (lane & 15);
      bf16x8 b0 = *(const bf16x8*)&Bs[n * 40 + bkb];
      bf16x8 b1 = *(const bf16x8*)&Bs[n * 40 + bkb + 16];
      bf16x16 bfv;
#pragma unroll
      for (int i = 0; i < 8; ++i) { bfv[i] = b0[i]; bfv[i + 8] = b1[i]; }
      acc[nb] = __builtin_amdgcn_wmma_f32_16x16x32_bf16(
          false, af, false, bfv, (short)0, acc[nb], false, false);
    }
    __syncthreads();
  }

  // epilogue: VGPR r -> row r (lanes 0-15) / r+8 (lanes 16-31); col = lane&15
  const int rbase = brow + wave * 16 + ((lane >> 4) ? 8 : 0);
  const int ncol = lane & 15;
  if (fullM) {
#pragma unroll
    for (int nb = 0; nb < 4; ++nb) {
      int gc = bcol + nb * 16 + ncol;
      float bv = bias ? bias[gc] : 0.0f;
#pragma unroll
      for (int r = 0; r < 8; ++r) {
        float v = acc[nb][r] + bv;
        if (mode & 1) v = v > 0.0f ? v : LRELU * v;
        long idx = (long)(rbase + r) * ldc + gc;
        if (mode & 2) v += C[idx];
        C[idx] = v;
      }
    }
  } else {
#pragma unroll
    for (int nb = 0; nb < 4; ++nb) {
      int gc = bcol + nb * 16 + ncol;
      float bv = bias ? bias[gc] : 0.0f;
#pragma unroll
      for (int r = 0; r < 8; ++r) {
        int gr = rbase + r;
        if (gr < M) {
          float v = acc[nb][r] + bv;
          if (mode & 1) v = v > 0.0f ? v : LRELU * v;
          long idx = (long)gr * ldc + gc;
          if (mode & 2) v += C[idx];
          C[idx] = v;
        }
      }
    }
  }
}

// ---------------------------------------------------------------------------
// Elementwise / segment kernels (vectorized where hot)
// ---------------------------------------------------------------------------
#define GTID long t = blockIdx.x * (long)blockDim.x + threadIdx.x

__global__ void k_zero(float* p, long n) { GTID; if (t < n) p[t] = 0.0f; }
__global__ void k_zero4(float4* p, long n4) {
  GTID; if (t < n4) p[t] = make_float4(0.f, 0.f, 0.f, 0.f);
}
__global__ void k_fill(float* p, long n, float v) { GTID; if (t < n) p[t] = v; }
__global__ void k_copy4(float4* d, const float4* s, long n4) {
  GTID; if (t < n4) d[t] = s[t];
}
__global__ void k_add4(float4* d, const float4* s, long n4) {
  GTID;
  if (t < n4) {
    float4 a = d[t], b = s[t];
    d[t] = make_float4(a.x + b.x, a.y + b.y, a.z + b.z, a.w + b.w);
  }
}

// copy [rows x 64] into column slice of [rows x ldd] (vec4)
__global__ void k_copy_strided4(float* d, int ldd, int dcol, const float* s,
                                int lds_, int rows) {
  GTID; long n = (long)rows * 16;
  if (t < n) {
    long r = t >> 4; int c = (int)(t & 15) * 4;
    *(float4*)&d[r * ldd + dcol + c] = *(const float4*)&s[r * lds_ + c];
  }
}

__global__ void k_build_edges(const int* ei, const int* batch, int* src, int* dst,
                              int nE, int N, int NA) {
  int i = blockIdx.x * blockDim.x + threadIdx.x;
  if (i < nE) { src[i] = ei[i]; dst[i] = ei[nE + i]; }
  if (i < N) {
    src[nE + i] = i;               dst[nE + i] = batch[i] + N;
    src[nE + N + i] = batch[i] + N; dst[nE + N + i] = i;
  }
  if (i < NA) { src[nE + 2 * N + i] = i; dst[nE + 2 * N + i] = i; }
}

__global__ void k_aug_batch(const int* batch, int* ab, int N, int NA) {
  int i = blockIdx.x * blockDim.x + threadIdx.x;
  if (i < NA) ab[i] = (i < N) ? batch[i] : (i - N);
}

__global__ void k_ep_fill4(float* auge, int nE, int twoN) {
  GTID; long n = (long)twoN * 16;
  if (t < n) {
    int d = (int)(t & 15) * 4;
    float4 v = make_float4(d == 0 ? 1.0f : 0.0f, 0.f, 0.f, 0.f);
    *(float4*)&auge[(long)nE * 64 + t * 4] = v;
  }
}

__global__ void k_loop_accum4(const float* auge, const int* dst, float* lsum, long n16) {
  GTID;
  if (t < n16) {
    long e = t >> 4; int d = (int)(t & 15) * 4;
    float4 v = *(const float4*)&auge[e * 64 + d];
    float* p = &lsum[(long)dst[e] * 64 + d];
    atomicAdd(p + 0, v.x); atomicAdd(p + 1, v.y);
    atomicAdd(p + 2, v.z); atomicAdd(p + 3, v.w);
  }
}
__global__ void k_loop_cnt(const int* dst, float* cnt, int n) {
  int i = blockIdx.x * blockDim.x + threadIdx.x;
  if (i < n) atomicAdd(&cnt[dst[i]], 1.0f);
}
__global__ void k_loop_write4(const float* lsum, const float* cnt, float* auge,
                              int base, long n16) {
  GTID;
  if (t < n16) {
    long r = t >> 4; int d = (int)(t & 15) * 4;
    float inv = 1.0f / fmaxf(cnt[r], 1.0f);
    float4 v = *(const float4*)&lsum[r * 64 + d];
    *(float4*)&auge[((long)base + r) * 64 + d] =
        make_float4(v.x * inv, v.y * inv, v.z * inv, v.w * inv);
  }
}

// graph layer-norm (scalar mean/var per graph over rows*64 values)
__global__ void k_ln_stats(const float* x, const int* seg, float* gsum,
                           float* gsq, float* gcnt, int rows) {
  int r = blockIdx.x * blockDim.x + threadIdx.x;
  if (r < rows) {
    const float4* p = (const float4*)(x + (long)r * 64);
    float s = 0.0f, q = 0.0f;
#pragma unroll
    for (int d = 0; d < 16; ++d) {
      float4 v = p[d];
      s += v.x + v.y + v.z + v.w;
      q += v.x * v.x + v.y * v.y + v.z * v.z + v.w * v.w;
    }
    int g = seg[r];
    atomicAdd(&gsum[g], s); atomicAdd(&gsq[g], q); atomicAdd(&gcnt[g], 1.0f);
  }
}
__global__ void k_ln_fin(float* gsum, float* gsq, const float* gcnt, int G) {
  int g = blockIdx.x * blockDim.x + threadIdx.x;
  if (g < G) {
    float c = fmaxf(gcnt[g], 1.0f) * 64.0f;
    float m = gsum[g] / c;
    float v = gsq[g] / c - m * m;
    gsum[g] = m;
    gsq[g] = rsqrtf(v + 1e-5f);
  }
}
__global__ void k_ln_apply4(const float* x, const int* seg, const float* mean,
                            const float* inv, float* out, long n16) {
  GTID;
  if (t < n16) {
    long r = t >> 4; int g = seg[r];
    float m = mean[g], iv = inv[g];
    float4 v = *(const float4*)&x[t * 4];
    *(float4*)&out[t * 4] =
        make_float4((v.x - m) * iv, (v.y - m) * iv, (v.z - m) * iv, (v.w - m) * iv);
  }
}

// message passing: agg[dst] += relu(onorm[src] + auge) + 1e-7
__global__ void k_msg4(const float* onrm, const float* auge, const int* src,
                       const int* dst, float* agg, long n16) {
  GTID;
  if (t < n16) {
    long e = t >> 4; int d = (int)(t & 15) * 4;
    float4 a = *(const float4*)&onrm[(long)src[e] * 64 + d];
    float4 b = *(const float4*)&auge[e * 64 + d];
    float* p = &agg[(long)dst[e] * 64 + d];
    atomicAdd(p + 0, fmaxf(a.x + b.x, 0.0f) + 1e-7f);
    atomicAdd(p + 1, fmaxf(a.y + b.y, 0.0f) + 1e-7f);
    atomicAdd(p + 2, fmaxf(a.z + b.z, 0.0f) + 1e-7f);
    atomicAdd(p + 3, fmaxf(a.w + b.w, 0.0f) + 1e-7f);
  }
}

// alpha[e,h] = 0.125 * dot(q[dst,h], k[src,h] + eh[e,h]); track segment max
__global__ void k_alpha(const float* q, const float* k, const float* eh,
                        const int* src, const int* dst, float* exb, float* amax,
                        long n) {
  GTID;
  if (t < n) {
    long e = t >> 1; int hd = (int)(t & 1);
    const float4* qp = (const float4*)(q + (long)dst[e] * 128 + hd * 64);
    const float4* kp = (const float4*)(k + (long)src[e] * 128 + hd * 64);
    const float4* ep = (const float4*)(eh + (long)e * 128 + hd * 64);
    float s = 0.0f;
#pragma unroll
    for (int d = 0; d < 16; ++d) {
      float4 a = qp[d], b = kp[d], c = ep[d];
      s += a.x * (b.x + c.x) + a.y * (b.y + c.y) +
           a.z * (b.z + c.z) + a.w * (b.w + c.w);
    }
    s *= 0.125f;
    exb[t] = s;
    atomicMaxF(&amax[(long)dst[e] * 2 + hd], s);
  }
}

__global__ void k_softmax_ed(float* exb, const int* dst, const float* amax,
                             float* den, long n) {
  GTID;
  if (t < n) {
    long e = t >> 1; int hd = (int)(t & 1);
    float x = __expf(exb[t] - amax[(long)dst[e] * 2 + hd]);
    exb[t] = x;
    atomicAdd(&den[(long)dst[e] * 2 + hd], x);
  }
}

__global__ void k_attn4(const float* v, const float* eh, const float* exb,
                        const float* den, const int* src, const int* dst,
                        float* atn, long n32) {
  GTID;
  if (t < n32) {
    long e = t >> 5; int d = (int)(t & 31) * 4; int hd = d >> 6;
    float a = exb[e * 2 + hd] / (den[(long)dst[e] * 2 + hd] + 1e-16f);
    float4 vv = *(const float4*)&v[(long)src[e] * 128 + d];
    float4 ee = *(const float4*)&eh[e * 128 + d];
    float* p = &atn[(long)dst[e] * 128 + d];
    atomicAdd(p + 0, (vv.x + ee.x) * a);
    atomicAdd(p + 1, (vv.y + ee.y) * a);
    atomicAdd(p + 2, (vv.z + ee.z) * a);
    atomicAdd(p + 3, (vv.w + ee.w) * a);
  }
}

__global__ void k_film4(float* o, const float* lh, const float* cs, const int* ab,
                        long n16) {
  GTID;
  if (t < n16) {
    long r = t >> 4; int d = (int)(t & 15) * 4; int g = ab[r];
    float4 sc = *(const float4*)&cs[(long)g * 128 + d];
    float4 sh = *(const float4*)&cs[(long)g * 128 + 64 + d];
    float4 ov = *(const float4*)&o[t * 4];
    float4 lv = *(const float4*)&lh[t * 4];
    *(float4*)&o[t * 4] = make_float4(ov.x + lv.x * sc.x + sh.x,
                                      ov.y + lv.y * sc.y + sh.y,
                                      ov.z + lv.z * sc.z + sh.z,
                                      ov.w + lv.w * sc.w + sh.w);
  }
}

__global__ void k_pool4(const float* o, const int* batch, float* pool, long n16) {
  GTID;
  if (t < n16) {
    long r = t >> 4; int d = (int)(t & 15) * 4;
    float4 v = *(const float4*)&o[t * 4];
    float* p = &pool[(long)batch[r] * 64 + d];
    atomicAdd(p + 0, v.x); atomicAdd(p + 1, v.y);
    atomicAdd(p + 2, v.z); atomicAdd(p + 3, v.w);
  }
}
__global__ void k_pcnt(const int* batch, float* pcnt, int N) {
  int i = blockIdx.x * blockDim.x + threadIdx.x;
  if (i < N) atomicAdd(&pcnt[batch[i]], 1.0f);
}
__global__ void k_glob(const float* pool, const float* pcnt, const float* o,
                       float* glob, int G, int N, long n) {
  GTID;
  if (t < n) {
    int g = (int)(t >> 7), d = (int)(t & 127);
    glob[t] = (d < 64) ? pool[(long)g * 64 + d] / fmaxf(pcnt[g], 1.0f)
                       : o[(long)(N + g) * 64 + (d - 64)];
  }
}

// ---------------------------------------------------------------------------
extern "C" void kernel_launch(void* const* d_in, const int* in_sizes, int n_in,
                              void* d_out, int out_size, void* d_ws, size_t ws_size,
                              hipStream_t stream) {
  const float* x     = (const float*)d_in[0];
  const float* eatt  = (const float*)d_in[1];
  const float* cond  = (const float*)d_in[2];
  const int*   eidx  = (const int*)d_in[3];
  const int*   batch = (const int*)d_in[4];

  const int N  = in_sizes[0] / 64;   // X_DIM = 64
  const int nE = in_sizes[1] / 16;   // E_DIM = 16
  const int G  = in_sizes[2] / 32;   // G_DIM = 32
  const int NA = N + G;
  const int NE6 = nE + 2 * N;
  const int NET = NE6 + NA;

  // ---- params: pytree flatten (dicts sorted by key, edge.b==None dropped) ----
  int pi = 5;
  auto P = [&]() -> const float* { return (const float*)d_in[pi++]; };
  const float *c2hW[3], *c2hB[3], *e2hW[3], *e2hB[3], *x2hW[3], *x2hB[3];
  const float *finW[2], *finB[2];
  struct Lyr {
    const float *csW, *csB, *edW, *ff0W, *ff0B, *ff1W, *ff1B, *genW, *genB,
                *kW, *kB, *linW, *linB, *qW, *qB, *skW, *skB, *vW, *vB;
  } L[3];
  for (int i = 0; i < 3; i++) { c2hW[i] = P(); c2hB[i] = P(); }
  for (int i = 0; i < 3; i++) { e2hW[i] = P(); e2hB[i] = P(); }
  for (int i = 0; i < 2; i++) { finW[i] = P(); finB[i] = P(); }
  for (int l = 0; l < 3; l++) {
    L[l].csW = P();  L[l].csB = P();
    L[l].edW = P();
    L[l].ff0W = P(); L[l].ff0B = P(); L[l].ff1W = P(); L[l].ff1B = P();
    L[l].genW = P(); L[l].genB = P();
    L[l].kW = P();   L[l].kB = P();
    L[l].linW = P(); L[l].linB = P();
    L[l].qW = P();   L[l].qB = P();
    L[l].skW = P();  L[l].skB = P();
    L[l].vW = P();   L[l].vB = P();
  }
  for (int i = 0; i < 3; i++) { x2hW[i] = P(); x2hB[i] = P(); }
  (void)n_in; (void)ws_size; (void)out_size;

  // ---- workspace carve ----
  char* wp = (char*)d_ws;
  auto alloc = [&](size_t bytes) -> void* {
    void* p = (void*)wp; wp += (bytes + 255) & ~(size_t)255; return p;
  };
  float* o    = (float*)alloc((size_t)NA * 64 * 4);
  float* onrm = (float*)alloc((size_t)NA * 64 * 4);
  float* agg  = (float*)alloc((size_t)NA * 64 * 4);
  float* lh   = (float*)alloc((size_t)NA * 64 * 4);
  float* hb   = (float*)alloc((size_t)NA * 128 * 4);
  float* qb   = (float*)alloc((size_t)NA * 128 * 4);
  float* kb   = (float*)alloc((size_t)NA * 128 * 4);
  float* vb   = (float*)alloc((size_t)NA * 128 * 4);
  float* atn  = (float*)alloc((size_t)NA * 128 * 4);
  float* ffh  = (float*)alloc((size_t)NA * 256 * 4);
  float* eh   = (float*)alloc((size_t)NET * 128 * 4);
  float* auge = (float*)alloc((size_t)NET * 64 * 4);
  float* exb  = (float*)alloc((size_t)NET * 2 * 4);
  float* amax = (float*)alloc((size_t)NA * 2 * 4);
  float* den  = (float*)alloc((size_t)NA * 2 * 4);
  float* cs   = (float*)alloc((size_t)G * 128 * 4);
  float* cemb = (float*)alloc((size_t)G * 64 * 4);
  float* gs1  = (float*)alloc((size_t)G * 4);
  float* gs2  = (float*)alloc((size_t)G * 4);
  float* gcv  = (float*)alloc((size_t)G * 4);
  float* pool = (float*)alloc((size_t)G * 64 * 4);
  float* pcnt = (float*)alloc((size_t)G * 4);
  float* glob = (float*)alloc((size_t)G * 128 * 4);
  int* srcA = (int*)alloc((size_t)NET * 4);
  int* dstA = (int*)alloc((size_t)NET * 4);
  int* ab   = (int*)alloc((size_t)NA * 4);

  auto cdiv = [](long a, long b) { return (a + b - 1) / b; };
#define EL(kern, n, ...) \
  kern<<<dim3((unsigned)cdiv((long)(n), 256)), dim3(256), 0, stream>>>(__VA_ARGS__)
  auto gemm = [&](const float* A, const float* W, const float* B, float* C,
                  int M, int K, int Nc, int ldc, int mode) {
    k_gemm<<<dim3((unsigned)cdiv(M, 64), (unsigned)(Nc / 64)), dim3(128), 0,
             stream>>>(A, W, B, C, M, K, Nc, ldc, mode);
  };

  // ---- input MLPs ----
  gemm(x,    x2hW[0], x2hB[0], onrm, N, 64, 64, 64, 1);
  gemm(onrm, x2hW[1], x2hB[1], agg,  N, 64, 64, 64, 1);
  gemm(agg,  x2hW[2], x2hB[2], o,    N, 64, 64, 64, 0);

  gemm(cond, c2hW[0], c2hB[0], pool, G, 32, 64, 64, 1);
  gemm(pool, c2hW[1], c2hB[1], glob, G, 64, 64, 64, 1);
  gemm(glob, c2hW[2], c2hB[2], cemb, G, 64, 64, 64, 0);
  EL(k_copy4, (long)G * 16, (float4*)(o + (size_t)N * 64), (const float4*)cemb,
     (long)G * 16);

  float* t1 = eh; float* t2 = eh + (size_t)nE * 64;
  gemm(eatt, e2hW[0], e2hB[0], t1,   nE, 16, 64, 64, 1);
  gemm(t1,   e2hW[1], e2hB[1], t2,   nE, 64, 64, 64, 1);
  gemm(t2,   e2hW[2], e2hB[2], auge, nE, 64, 64, 64, 0);

  // ---- augmented graph structure ----
  EL(k_build_edges, nE, eidx, batch, srcA, dstA, nE, N, NA);
  EL(k_aug_batch, NA, batch, ab, N, NA);
  EL(k_ep_fill4, (long)2 * N * 16, auge, nE, 2 * N);
  EL(k_zero4, (long)NA * 16, (float4*)lh, (long)NA * 16);   // loop-attr sum
  EL(k_zero, NA, amax, (long)NA);                           // in-degree count
  EL(k_loop_accum4, (long)NE6 * 16, auge, dstA, lh, (long)NE6 * 16);
  EL(k_loop_cnt, NE6, dstA, amax, NE6);
  EL(k_loop_write4, (long)NA * 16, lh, amax, auge, NE6, (long)NA * 16);

  // ---- transformer layers ----
  for (int l = 0; l < 3; ++l) {
    const Lyr& y = L[l];
    gemm(cemb, y.csW, y.csB, cs, G, 64, 128, 128, 0);
    // graph LN -> onrm
    EL(k_zero, G, gs1, (long)G); EL(k_zero, G, gs2, (long)G); EL(k_zero, G, gcv, (long)G);
    EL(k_ln_stats, NA, o, ab, gs1, gs2, gcv, NA);
    EL(k_ln_fin, G, gs1, gs2, gcv, G);
    EL(k_ln_apply4, (long)NA * 16, o, ab, gs1, gs2, onrm, (long)NA * 16);
    // message aggregation
    EL(k_zero4, (long)NA * 16, (float4*)agg, (long)NA * 16);
    EL(k_msg4, (long)NET * 16, onrm, auge, srcA, dstA, agg, (long)NET * 16);
    EL(k_add4, (long)NA * 16, (float4*)agg, (const float4*)onrm, (long)NA * 16);
    gemm(agg, y.genW, y.genB, hb + 64, NA, 64, 64, 128, 0);
    EL(k_copy_strided4, (long)NA * 16, hb, 128, 0, onrm, 64, NA);
    // q/k/v/skip + edge projections
    gemm(hb, y.qW, y.qB, qb, NA, 128, 128, 128, 0);
    gemm(hb, y.kW, y.kB, kb, NA, 128, 128, 128, 0);
    gemm(hb, y.vW, y.vB, vb, NA, 128, 128, 128, 0);
    gemm(hb, y.skW, y.skB, atn, NA, 128, 128, 128, 0);  // skip pre-fills attn
    gemm(auge, y.edW, nullptr, eh, NET, 64, 128, 128, 0);
    // segment softmax attention
    EL(k_fill, (long)NA * 2, amax, (long)NA * 2, -3.0e38f);
    EL(k_zero, (long)NA * 2, den, (long)NA * 2);
    EL(k_alpha, (long)NET * 2, qb, kb, eh, srcA, dstA, exb, amax, (long)NET * 2);
    EL(k_softmax_ed, (long)NET * 2, exb, dstA, amax, den, (long)NET * 2);
    EL(k_attn4, (long)NET * 32, vb, eh, exb, den, srcA, dstA, atn, (long)NET * 32);
    gemm(atn, y.linW, y.linB, lh, NA, 128, 64, 64, 0);
    EL(k_film4, (long)NA * 16, o, lh, cs, ab, (long)NA * 16);
    // feed-forward with second graph LN
    EL(k_zero, G, gs1, (long)G); EL(k_zero, G, gs2, (long)G); EL(k_zero, G, gcv, (long)G);
    EL(k_ln_stats, NA, o, ab, gs1, gs2, gcv, NA);
    EL(k_ln_fin, G, gs1, gs2, gcv, G);
    EL(k_ln_apply4, (long)NA * 16, o, ab, gs1, gs2, onrm, (long)NA * 16);
    gemm(onrm, y.ff0W, y.ff0B, ffh, NA, 64, 256, 256, 1);
    gemm(ffh, y.ff1W, y.ff1B, o, NA, 256, 64, 64, 2);   // o += ff
  }

  // ---- readout ----
  EL(k_zero4, (long)G * 16, (float4*)pool, (long)G * 16);
  EL(k_zero, G, pcnt, (long)G);
  EL(k_pool4, (long)N * 16, o, batch, pool, (long)N * 16);
  EL(k_pcnt, N, batch, pcnt, N);
  EL(k_glob, (long)G * 128, pool, pcnt, o, glob, G, N, (long)G * 128);
  gemm(glob, finW[0], finB[0], pool, G, 128, 64, 64, 1);
  gemm(pool, finW[1], finB[1], (float*)d_out, G, 64, 64, 64, 0);
#undef EL
}
